// LSTMModel_1425929142284
// MI455X (gfx1250) — compile-verified, matrix operand order checked by reading, not run on hardware
//
#include <hip/hip_runtime.h>
#include <hip/hip_bf16.h>

// ---------------------------------------------------------------------------
// LSTM on MI455X (gfx1250): bf16 WMMA (v_wmma_f32_16x16x32_bf16) everywhere.
//   B=64, T=256, D=512, H=1024, O=256
// ---------------------------------------------------------------------------

#define B_ 64
#define T_ 256
#define D_ 512
#define H_ 1024
#define G4H_ 4096   // 4*H
#define O_ 256

typedef __attribute__((ext_vector_type(16))) __bf16 bf16x16;
typedef __attribute__((ext_vector_type(8)))  float  f32x8;

union Frag16 {
    bf16x16 v;
    uint4   q[2];
};

// Load one 16x32 bf16 A/B fragment from a row-major [rows x ld] matrix.
// row_base must already point at (base + row*ld). Per the CDNA5 ISA layout:
// lanes 0-15: K = k0+0..7 and k0+16..23 ; lanes 16-31: K = k0+8..15 and k0+24..31.
// Both chunks are contiguous 16-byte loads.
__device__ __forceinline__ bf16x16 load_frag(const __hip_bfloat16* row_base,
                                             int k0, int lane) {
    const int kbase = (lane & 16) ? 8 : 0;
    const uint4* p = reinterpret_cast<const uint4*>(row_base + k0 + kbase);
    Frag16 f;
    f.q[0] = p[0];   // 8 bf16 at k0+kbase
    f.q[1] = p[2];   // 8 bf16 at k0+kbase+16
    return f.v;
}

__device__ __forceinline__ float sigmoid_(float x) {
    return 1.0f / (1.0f + __expf(-x));
}

// ---------------------------------------------------------------------------
// fp32 -> bf16 conversion
// ---------------------------------------------------------------------------
__global__ void __launch_bounds__(256) cvt_f32_bf16_kernel(
    const float* __restrict__ in, __hip_bfloat16* __restrict__ out, int n) {
    int i = blockIdx.x * 256 + threadIdx.x;
    if (i < n) out[i] = __float2bfloat16(in[i]);
}

// Zero h (bf16 ping buffer 0) and c (fp32). 64*1024 = 65536 elements each.
__global__ void __launch_bounds__(256) init_state_kernel(
    __hip_bfloat16* __restrict__ h0, float* __restrict__ c) {
    int i = blockIdx.x * 256 + threadIdx.x;
    if (i < B_ * H_) {
        h0[i] = __float2bfloat16(0.0f);
        c[i]  = 0.0f;
    }
}

// ---------------------------------------------------------------------------
// Input projection: xg[m][g] = sum_d x[m][d] * W_ih[g][d] + b_ih[g] + b_hh[g]
//   m = b*T + t, M=16384, N=4096, K=512.
// Block: 256 thr (8 waves). Wave w: m_tile = w&3, n_sub = w>>2.
// Grid: (4096/32, 16384/64). Each wave: one 16x16 tile, 16 WMMAs over K.
// ---------------------------------------------------------------------------
__global__ void __launch_bounds__(256) proj_kernel(
    const __hip_bfloat16* __restrict__ xb,   // [16384, 512]
    const __hip_bfloat16* __restrict__ Wb,   // [4096, 512]
    const float* __restrict__ b_ih,
    const float* __restrict__ b_hh,
    float* __restrict__ xg)                  // [16384, 4096]
{
    const int tid  = threadIdx.x;
    const int lane = tid & 31;
    const int w    = tid >> 5;
    const int m0   = blockIdx.y * 64 + (w & 3) * 16;
    const int n0   = blockIdx.x * 32 + (w >> 2) * 16;

    const int a_row = m0 + (lane & 15);
    const int n_col = n0 + (lane & 15);
    const __hip_bfloat16* a_base = xb + (size_t)a_row * D_;
    const __hip_bfloat16* b_base = Wb + (size_t)n_col * D_;

    f32x8 acc = {};
    for (int k0 = 0; k0 < D_; k0 += 32) {
        bf16x16 a = load_frag(a_base, k0, lane);
        bf16x16 b = load_frag(b_base, k0, lane);
        acc = __builtin_amdgcn_wmma_f32_16x16x32_bf16(
            false, a, false, b, (short)0, acc, false, false);
    }

    const float bias = b_ih[n_col] + b_hh[n_col];
#pragma unroll
    for (int r = 0; r < 8; ++r) {
        const int m = m0 + r + ((lane >> 4) << 3);
        xg[(size_t)m * G4H_ + n_col] = acc[r] + bias;
    }
}

// ---------------------------------------------------------------------------
// One LSTM timestep: gates = xg[:,t,:] + h_in @ W_hh^T ; cell update.
// Block: 256 thr (8 waves), covers all 64 batch rows x 32 hidden cols,
// all 4 gates (wave keeps 4 accumulators, reuses the A fragment 4x).
// Grid: H/32 = 32 blocks. h ping-pongs across launches (no races).
// ---------------------------------------------------------------------------
__global__ void __launch_bounds__(256) lstm_step_kernel(
    const float* __restrict__ xg,            // [B*T, 4H]
    const __hip_bfloat16* __restrict__ Whh,  // [4H, H]
    const __hip_bfloat16* __restrict__ h_in, // [B, H]
    float* __restrict__ c,                   // [B, H]
    __hip_bfloat16* __restrict__ h_out,      // [B, H]
    int t)
{
    const int tid  = threadIdx.x;
    const int lane = tid & 31;
    const int w    = tid >> 5;
    const int m0   = (w & 3) * 16;                    // batch tile
    const int j0   = blockIdx.x * 32 + (w >> 2) * 16; // hidden col tile

    const int n_col = j0 + (lane & 15);               // hidden column
    const __hip_bfloat16* a_base = h_in + (size_t)(m0 + (lane & 15)) * H_;

    f32x8 acc[4] = {{}, {}, {}, {}};                  // i, f, g, o

    for (int k0 = 0; k0 < H_; k0 += 32) {
        bf16x16 a = load_frag(a_base, k0, lane);
#pragma unroll
        for (int g = 0; g < 4; ++g) {
            const __hip_bfloat16* b_base =
                Whh + (size_t)(g * H_ + n_col) * H_;
            bf16x16 b = load_frag(b_base, k0, lane);
            acc[g] = __builtin_amdgcn_wmma_f32_16x16x32_bf16(
                false, a, false, b, (short)0, acc[g], false, false);
        }
    }

#pragma unroll
    for (int r = 0; r < 8; ++r) {
        const int m = m0 + r + ((lane >> 4) << 3);    // batch index
        const float* xg_row = xg + ((size_t)m * T_ + t) * G4H_;
        const float gi = acc[0][r] + xg_row[n_col];
        const float gf = acc[1][r] + xg_row[n_col + H_];
        const float gg = acc[2][r] + xg_row[n_col + 2 * H_];
        const float go = acc[3][r] + xg_row[n_col + 3 * H_];

        const size_t idx = (size_t)m * H_ + n_col;
        const float cn = sigmoid_(gf) * c[idx] + sigmoid_(gi) * tanhf(gg);
        c[idx] = cn;
        h_out[idx] = __float2bfloat16(sigmoid_(go) * tanhf(cn));
    }
}

// ---------------------------------------------------------------------------
// Final FC: out[b][o] = sum_j h[b][j] * W_fc[o][j] + b_fc[o]
//   M=64, N=256, K=1024. Grid: 256/32 = 8 blocks.
// ---------------------------------------------------------------------------
__global__ void __launch_bounds__(256) fc_kernel(
    const __hip_bfloat16* __restrict__ h,    // [64, 1024]
    const __hip_bfloat16* __restrict__ Wb,   // [256, 1024]
    const float* __restrict__ b_fc,
    float* __restrict__ out)                 // [64, 256]
{
    const int tid  = threadIdx.x;
    const int lane = tid & 31;
    const int w    = tid >> 5;
    const int m0   = (w & 3) * 16;
    const int n0   = blockIdx.x * 32 + (w >> 2) * 16;

    const int n_col = n0 + (lane & 15);
    const __hip_bfloat16* a_base = h  + (size_t)(m0 + (lane & 15)) * H_;
    const __hip_bfloat16* b_base = Wb + (size_t)n_col * H_;

    f32x8 acc = {};
    for (int k0 = 0; k0 < H_; k0 += 32) {
        bf16x16 a = load_frag(a_base, k0, lane);
        bf16x16 b = load_frag(b_base, k0, lane);
        acc = __builtin_amdgcn_wmma_f32_16x16x32_bf16(
            false, a, false, b, (short)0, acc, false, false);
    }

    const float bias = b_fc[n_col];
#pragma unroll
    for (int r = 0; r < 8; ++r) {
        const int m = m0 + r + ((lane >> 4) << 3);
        out[(size_t)m * O_ + n_col] = acc[r] + bias;
    }
}

// ---------------------------------------------------------------------------
// Host-side launcher (graph-capture safe: launches only, all on `stream`)
// ---------------------------------------------------------------------------
extern "C" void kernel_launch(void* const* d_in, const int* in_sizes, int n_in,
                              void* d_out, int out_size, void* d_ws, size_t ws_size,
                              hipStream_t stream) {
    (void)in_sizes; (void)n_in; (void)out_size; (void)ws_size;

    const float* x    = (const float*)d_in[0];   // [B,T,D]
    const float* W_ih = (const float*)d_in[1];   // [4H,D]
    const float* W_hh = (const float*)d_in[2];   // [4H,H]
    const float* b_ih = (const float*)d_in[3];   // [4H]
    const float* b_hh = (const float*)d_in[4];   // [4H]
    const float* W_fc = (const float*)d_in[5];   // [O,H]
    const float* b_fc = (const float*)d_in[6];   // [O]
    float* out = (float*)d_out;                  // [B,O]

    // Workspace carve-up (all offsets 256B-aligned by construction).
    char* p = (char*)d_ws;
    float* xg = (float*)p;                        p += (size_t)B_ * T_ * G4H_ * 4; // 268.4 MB
    __hip_bfloat16* x_b   = (__hip_bfloat16*)p;   p += (size_t)B_ * T_ * D_ * 2;   // 16.8 MB
    __hip_bfloat16* Wih_b = (__hip_bfloat16*)p;   p += (size_t)G4H_ * D_ * 2;      // 4 MB
    __hip_bfloat16* Whh_b = (__hip_bfloat16*)p;   p += (size_t)G4H_ * H_ * 2;      // 8 MB
    __hip_bfloat16* Wfc_b = (__hip_bfloat16*)p;   p += (size_t)O_ * H_ * 2;        // 0.5 MB
    __hip_bfloat16* hbuf0 = (__hip_bfloat16*)p;   p += (size_t)B_ * H_ * 2;
    __hip_bfloat16* hbuf1 = (__hip_bfloat16*)p;   p += (size_t)B_ * H_ * 2;
    float* c = (float*)p;                         p += (size_t)B_ * H_ * 4;

    // 1) Convert fp32 inputs/weights to bf16.
    {
        int n;
        n = B_ * T_ * D_;
        cvt_f32_bf16_kernel<<<(n + 255) / 256, 256, 0, stream>>>(x, x_b, n);
        n = G4H_ * D_;
        cvt_f32_bf16_kernel<<<(n + 255) / 256, 256, 0, stream>>>(W_ih, Wih_b, n);
        n = G4H_ * H_;
        cvt_f32_bf16_kernel<<<(n + 255) / 256, 256, 0, stream>>>(W_hh, Whh_b, n);
        n = O_ * H_;
        cvt_f32_bf16_kernel<<<(n + 255) / 256, 256, 0, stream>>>(W_fc, Wfc_b, n);
    }

    // 2) Zero h (ping buffer 0) and c.
    init_state_kernel<<<(B_ * H_ + 255) / 256, 256, 0, stream>>>(hbuf0, c);

    // 3) Input projection GEMM (WMMA): xg = x @ W_ih^T + b_ih + b_hh.
    {
        dim3 grid(G4H_ / 32, (B_ * T_) / 64);
        proj_kernel<<<grid, 256, 0, stream>>>(x_b, Wih_b, b_ih, b_hh, xg);
    }

    // 4) Sequential recurrence: 256 step launches, h ping-pongs.
    for (int t = 0; t < T_; ++t) {
        const __hip_bfloat16* h_in  = (t & 1) ? hbuf1 : hbuf0;
        __hip_bfloat16*       h_out = (t & 1) ? hbuf0 : hbuf1;
        lstm_step_kernel<<<H_ / 32, 256, 0, stream>>>(xg, Whh_b, h_in, c, h_out, t);
    }
    // After t=255 (odd), final h is in hbuf0.

    // 5) Final FC (WMMA).
    fc_kernel<<<O_ / 32, 256, 0, stream>>>(hbuf0, Wfc_b, b_fc, out);
}
